// DeformBasicBlock1_6313601925211
// MI455X (gfx1250) — matrix operand order repaired and checked by reading, not compiled, not run on hardware
//
#include <hip/hip_runtime.h>

// ---------------------------------------------------------------------------
// Problem constants (B,C,D,H,W) = (2,64,8,32,32), cpg=8, G=8
// ---------------------------------------------------------------------------
constexpr int CIN   = 64;
constexpr int DD    = 8;
constexpr int HH    = 32;
constexpr int WW    = 32;
constexpr int DHW   = DD * HH * WW;        // 8192
constexpr int NB    = 2;                   // batch
constexpr int NTOT  = NB * DHW;            // 16384  (GEMM N)
constexpr int KTOT  = CIN * 27;            // 1728   (GEMM K, multiple of 32)
constexpr int COFF  = 648;                 // offset-conv output channels
constexpr int MPAD  = 672;                 // COFF padded to multiple of 32
constexpr float BN_EPS = 1e-5f;

typedef __bf16 v16bf __attribute__((ext_vector_type(16)));
typedef float  v8f   __attribute__((ext_vector_type(8)));

union Frag {               // one WMMA operand fragment (16 bf16 = 8 VGPRs/lane)
  v16bf  v;
  uint4  q[2];
  unsigned u[8];
  __bf16 h[16];
};

__device__ __forceinline__ void split_bf16(float x, __bf16& hi, __bf16& lo) {
  __bf16 h = (__bf16)x;                    // RNE high part
  hi = h;
  lo = (__bf16)(x - (float)h);             // residual -> ~fp32 via 3x bf16 WMMA
}

// ---------------------------------------------------------------------------
// im2col of x (3x3x3, pad 1) into bf16 hi/lo planes, layout col[n][k], k fast.
// ---------------------------------------------------------------------------
__global__ __launch_bounds__(256)
void im2col_x_kernel(const float* __restrict__ x,
                     __bf16* __restrict__ colHi, __bf16* __restrict__ colLo) {
  size_t tid = (size_t)blockIdx.x * 256 + threadIdx.x;
  if (tid >= (size_t)NTOT * KTOT) return;
  int k = (int)(tid % KTOT);
  int n = (int)(tid / KTOT);
  int ci = k / 27, kidx = k % 27;
  int kd = kidx / 9 - 1, kh = (kidx / 3) % 3 - 1, kw = kidx % 3 - 1;
  int b = n >> 13, s = n & (DHW - 1);
  int d = s >> 10, h = (s >> 5) & 31, w = s & 31;
  int dz = d + kd, hz = h + kh, wz = w + kw;
  float v = 0.0f;
  if ((unsigned)dz < (unsigned)DD && (unsigned)hz < (unsigned)HH &&
      (unsigned)wz < (unsigned)WW)
    v = x[((size_t)b * CIN + ci) * DHW + ((dz << 5) + hz) * 32 + wz];
  __bf16 hi, lo; split_bf16(v, hi, lo);
  colHi[tid] = hi;                         // tid == n*KTOT + k
  colLo[tid] = lo;
}

// ---------------------------------------------------------------------------
// Split fp32 weights (Mreal x KTOT) into zero-padded bf16 hi/lo (Mpad x KTOT).
// ---------------------------------------------------------------------------
__global__ __launch_bounds__(256)
void split_w_kernel(const float* __restrict__ w,
                    __bf16* __restrict__ hi, __bf16* __restrict__ lo,
                    int Mreal, int Mpad) {
  size_t tid = (size_t)blockIdx.x * 256 + threadIdx.x;
  if (tid >= (size_t)Mpad * KTOT) return;
  int m = (int)(tid / KTOT);
  float v = 0.0f;
  if (m < Mreal) v = w[tid];
  __bf16 h, l; split_bf16(v, h, l);
  hi[tid] = h; lo[tid] = l;
}

// ---------------------------------------------------------------------------
// WMMA GEMM: out[b][m][s] = sum_k A[m][k] * col[n][k] (+bias[m]),
// emulated fp32 = A_hi*B_hi + A_hi*B_lo + A_lo*B_hi with f32 accumulate.
// One wave (32 threads) per 32(M) x 32(N) tile => 12 WMMAs per 32-wide
// K-step against one A/B LDS staging pass (2x the matrix-op density of a
// 16x32 tile). A/B slabs staged via b128 LDS ops; next slab prefetched with
// global_prefetch_b8.
// ---------------------------------------------------------------------------
__global__ __launch_bounds__(32)
void wmma_gemm_kernel(const __bf16* __restrict__ aHi, const __bf16* __restrict__ aLo,
                      const __bf16* __restrict__ colHi, const __bf16* __restrict__ colLo,
                      const float* __restrict__ bias,
                      float* __restrict__ out, int Mreal) {
  __shared__ __bf16 ldsAH[32][32] __attribute__((aligned(16)));
  __shared__ __bf16 ldsAL[32][32] __attribute__((aligned(16)));
  __shared__ __bf16 ldsBH[32][32] __attribute__((aligned(16)));
  __shared__ __bf16 ldsBL[32][32] __attribute__((aligned(16)));

  const int lane    = threadIdx.x;
  const int mlane   = lane & 15;
  const int halfSel = lane >> 4;           // wave halves per ISA VGPR layout
  const int mTile   = blockIdx.x * 32;
  const int n0      = blockIdx.y * 32;

  v8f c00 = {};                            // (M 0..15 , N 0..15)
  v8f c01 = {};                            // (M 0..15 , N 16..31)
  v8f c10 = {};                            // (M 16..31, N 0..15)
  v8f c11 = {};                            // (M 16..31, N 16..31)

  for (int k0 = 0; k0 < KTOT; k0 += 32) {
    // ---- stage 32 A rows (hi+lo) and 32 B rows (hi+lo): 4 rows per lane
    {
      const uint4* sAH = (const uint4*)(aHi   + (size_t)(mTile + lane) * KTOT + k0);
      const uint4* sAL = (const uint4*)(aLo   + (size_t)(mTile + lane) * KTOT + k0);
      const uint4* sBH = (const uint4*)(colHi + (size_t)(n0    + lane) * KTOT + k0);
      const uint4* sBL = (const uint4*)(colLo + (size_t)(n0    + lane) * KTOT + k0);
      uint4* dAH = (uint4*)&ldsAH[lane][0];
      uint4* dAL = (uint4*)&ldsAL[lane][0];
      uint4* dBH = (uint4*)&ldsBH[lane][0];
      uint4* dBL = (uint4*)&ldsBL[lane][0];
#pragma unroll
      for (int i = 0; i < 4; ++i) {
        dAH[i] = sAH[i];
        dAL[i] = sAL[i];
        dBH[i] = sBH[i];
        dBL[i] = sBL[i];
      }
    }
    if (k0 + 32 < KTOT) {                  // pull next K slab toward L2/L0
      __builtin_prefetch(colHi + (size_t)(n0 + lane) * KTOT + k0 + 32, 0, 0);
      __builtin_prefetch(colLo + (size_t)(n0 + lane) * KTOT + k0 + 32, 0, 0);
    }
    __syncthreads();

    // ---- A fragments (16-bit A 16x32 layout: K-pair index p per ISA table)
    Frag a0H, a0L, a1H, a1L;
    {
      const unsigned* r0H = (const unsigned*)&ldsAH[mlane][0];
      const unsigned* r0L = (const unsigned*)&ldsAL[mlane][0];
      const unsigned* r1H = (const unsigned*)&ldsAH[mlane + 16][0];
      const unsigned* r1L = (const unsigned*)&ldsAL[mlane + 16][0];
#pragma unroll
      for (int v = 0; v < 8; ++v) {
        int p = (v < 4 ? v : v + 4) + halfSel * 4;  // K pair index
        a0H.u[v] = r0H[p];
        a0L.u[v] = r0L[p];
        a1H.u[v] = r1H[p];
        a1L.u[v] = r1L[p];
      }
    }
    // ---- B fragments (B 32x16: lane half h holds K 8h..8h+7 and 16+8h..16+8h+7, matching A)
    Frag b0H, b0L, b1H, b1L;
    {
      const uint4* rB;
      rB = (const uint4*)&ldsBH[mlane     ][halfSel * 8]; b0H.q[0]=rB[0]; b0H.q[1]=rB[2];
      rB = (const uint4*)&ldsBL[mlane     ][halfSel * 8]; b0L.q[0]=rB[0]; b0L.q[1]=rB[2];
      rB = (const uint4*)&ldsBH[mlane + 16][halfSel * 8]; b1H.q[0]=rB[0]; b1H.q[1]=rB[2];
      rB = (const uint4*)&ldsBL[mlane + 16][halfSel * 8]; b1L.q[0]=rB[0]; b1L.q[1]=rB[2];
    }

    // ---- 3x bf16 WMMA per (M-tile, N-tile) = emulated fp32 MAC, 12 total
    c00 = __builtin_amdgcn_wmma_f32_16x16x32_bf16(false, a0H.v, false, b0H.v, (short)0, c00, false, false);
    c00 = __builtin_amdgcn_wmma_f32_16x16x32_bf16(false, a0H.v, false, b0L.v, (short)0, c00, false, false);
    c00 = __builtin_amdgcn_wmma_f32_16x16x32_bf16(false, a0L.v, false, b0H.v, (short)0, c00, false, false);

    c01 = __builtin_amdgcn_wmma_f32_16x16x32_bf16(false, a0H.v, false, b1H.v, (short)0, c01, false, false);
    c01 = __builtin_amdgcn_wmma_f32_16x16x32_bf16(false, a0H.v, false, b1L.v, (short)0, c01, false, false);
    c01 = __builtin_amdgcn_wmma_f32_16x16x32_bf16(false, a0L.v, false, b1H.v, (short)0, c01, false, false);

    c10 = __builtin_amdgcn_wmma_f32_16x16x32_bf16(false, a1H.v, false, b0H.v, (short)0, c10, false, false);
    c10 = __builtin_amdgcn_wmma_f32_16x16x32_bf16(false, a1H.v, false, b0L.v, (short)0, c10, false, false);
    c10 = __builtin_amdgcn_wmma_f32_16x16x32_bf16(false, a1L.v, false, b0H.v, (short)0, c10, false, false);

    c11 = __builtin_amdgcn_wmma_f32_16x16x32_bf16(false, a1H.v, false, b1H.v, (short)0, c11, false, false);
    c11 = __builtin_amdgcn_wmma_f32_16x16x32_bf16(false, a1H.v, false, b1L.v, (short)0, c11, false, false);
    c11 = __builtin_amdgcn_wmma_f32_16x16x32_bf16(false, a1L.v, false, b1H.v, (short)0, c11, false, false);
    __syncthreads();
  }

  // ---- store D (C/D layout: VGPR r -> M = r + halfSel*8, N = lane&15)
  const bool hasBias = (bias != nullptr);
  const int nA = n0 + mlane;
  const int nB = nA + 16;
  const int bA = nA >> 13, sA = nA & (DHW - 1);
  const int bB = nB >> 13, sB = nB & (DHW - 1);
#pragma unroll
  for (int r = 0; r < 8; ++r) {
    int m0 = mTile + r + halfSel * 8;      // M sub-tile 0
    int m1 = m0 + 16;                      // M sub-tile 1
    if (m0 < Mreal) {
      float bv = hasBias ? bias[m0] : 0.0f;
      out[((size_t)bA * Mreal + m0) * DHW + sA] = c00[r] + bv;
      out[((size_t)bB * Mreal + m0) * DHW + sB] = c01[r] + bv;
    }
    if (m1 < Mreal) {
      float bv = hasBias ? bias[m1] : 0.0f;
      out[((size_t)bA * Mreal + m1) * DHW + sA] = c10[r] + bv;
      out[((size_t)bB * Mreal + m1) * DHW + sB] = c11[r] + bv;
    }
  }
}

// ---------------------------------------------------------------------------
// Deformable im2col: one thread per (b,g,k27,s) computes trilinear interp
// once and gathers all 8 channels of the group (shares offset/corner math).
// ---------------------------------------------------------------------------
__global__ __launch_bounds__(256)
void deform_im2col_kernel(const float* __restrict__ src,
                          const float* __restrict__ off,
                          __bf16* __restrict__ colHi, __bf16* __restrict__ colLo) {
  size_t tid = (size_t)blockIdx.x * 256 + threadIdx.x;
  if (tid >= (size_t)NB * 8 * 27 * DHW) return;
  int s   = (int)(tid & (DHW - 1));
  int t   = (int)(tid >> 13);
  int k27 = t % 27; t /= 27;
  int g   = t & 7;
  int b   = t >> 3;

  int d = s >> 10, h = (s >> 5) & 31, w = s & 31;
  int kd = k27 / 9 - 1, kh = (k27 / 3) % 3 - 1, kw = k27 % 3 - 1;

  size_t ob = ((size_t)b * COFF + ((size_t)g * 27 + k27) * 3) * DHW + s;
  float pd = (float)(d + kd) + off[ob];
  float ph = (float)(h + kh) + off[ob + DHW];
  float pw = (float)(w + kw) + off[ob + 2 * DHW];

  float d0f = floorf(pd), h0f = floorf(ph), w0f = floorf(pw);
  float fd = pd - d0f, fh = ph - h0f, fw = pw - w0f;
  int d0 = (int)d0f, h0 = (int)h0f, w0 = (int)w0f;

  float acc[8] = {0, 0, 0, 0, 0, 0, 0, 0};
  const float* sb = src + ((size_t)b * CIN + g * 8) * DHW;
#pragma unroll
  for (int corner = 0; corner < 8; ++corner) {
    int dd = corner >> 2, hh2 = (corner >> 1) & 1, ww2 = corner & 1;
    int di = d0 + dd, hi_ = h0 + hh2, wi = w0 + ww2;
    if ((unsigned)di < (unsigned)DD && (unsigned)hi_ < (unsigned)HH &&
        (unsigned)wi < (unsigned)WW) {
      float wt = (dd ? fd : 1.0f - fd) * (hh2 ? fh : 1.0f - fh) *
                 (ww2 ? fw : 1.0f - fw);
      int idx = ((di << 5) + hi_) * 32 + wi;
#pragma unroll
      for (int c = 0; c < 8; ++c)
        acc[c] += wt * sb[(size_t)c * DHW + idx];
    }
  }
  int n = (b << 13) + s;
#pragma unroll
  for (int c = 0; c < 8; ++c) {
    size_t kk = (size_t)((g * 8 + c) * 27 + k27);
    __bf16 hi, lo; split_bf16(acc[c], hi, lo);
    colHi[(size_t)n * KTOT + kk] = hi;
    colLo[(size_t)n * KTOT + kk] = lo;
  }
}

// ---------------------------------------------------------------------------
// BatchNorm statistics: one block per channel, LDS tree reduction.
// stats[c] = mean, stats[64+c] = rsqrt(var + eps)
// ---------------------------------------------------------------------------
__global__ __launch_bounds__(256)
void bn_stats_kernel(const float* __restrict__ y, float* __restrict__ stats) {
  __shared__ float sSum[256];
  __shared__ float sSq[256];
  int c = blockIdx.x;
  float sum = 0.0f, sq = 0.0f;
  for (int i = threadIdx.x; i < NB * DHW; i += 256) {
    int b = i >> 13, s = i & (DHW - 1);
    float v = y[((size_t)b * CIN + c) * DHW + s];
    sum += v; sq += v * v;
  }
  sSum[threadIdx.x] = sum; sSq[threadIdx.x] = sq;
  __syncthreads();
  for (int stride = 128; stride > 0; stride >>= 1) {
    if (threadIdx.x < stride) {
      sSum[threadIdx.x] += sSum[threadIdx.x + stride];
      sSq[threadIdx.x]  += sSq[threadIdx.x + stride];
    }
    __syncthreads();
  }
  if (threadIdx.x == 0) {
    float inv = 1.0f / (float)(NB * DHW);
    float mu  = sSum[0] * inv;
    float var = sSq[0] * inv - mu * mu;
    stats[c]      = mu;
    stats[CIN + c] = rsqrtf(var + BN_EPS);
  }
}

// ---------------------------------------------------------------------------
// BN apply (+optional residual, +optional ReLU), elementwise.
// ---------------------------------------------------------------------------
__global__ __launch_bounds__(256)
void bn_apply_kernel(const float* __restrict__ y, const float* __restrict__ stats,
                     const float* __restrict__ gamma, const float* __restrict__ beta,
                     const float* __restrict__ residual, float* __restrict__ out,
                     int doRelu) {
  size_t tid = (size_t)blockIdx.x * 256 + threadIdx.x;
  if (tid >= (size_t)NB * CIN * DHW) return;
  int c = (int)((tid >> 13) & (CIN - 1));
  float v = gamma[c] * (y[tid] - stats[c]) * stats[CIN + c] + beta[c];
  if (residual) v += residual[tid];
  if (doRelu) v = fmaxf(v, 0.0f);
  out[tid] = v;
}

// ---------------------------------------------------------------------------
// Host-side orchestration (all launches on `stream`, graph-capture safe).
// ---------------------------------------------------------------------------
extern "C" void kernel_launch(void* const* d_in, const int* in_sizes, int n_in,
                              void* d_out, int out_size, void* d_ws, size_t ws_size,
                              hipStream_t stream) {
  const float* x      = (const float*)d_in[0];
  const float* w_off1 = (const float*)d_in[1];
  const float* b_off1 = (const float*)d_in[2];
  const float* w1     = (const float*)d_in[3];
  const float* g1     = (const float*)d_in[4];
  const float* be1    = (const float*)d_in[5];
  const float* w_off2 = (const float*)d_in[6];
  const float* b_off2 = (const float*)d_in[7];
  const float* w2     = (const float*)d_in[8];
  const float* g2     = (const float*)d_in[9];
  const float* be2    = (const float*)d_in[10];

  float* out  = (float*)d_out;                               // (B,64,DHW)
  float* off1 = out + (size_t)NB * CIN * DHW;                // (B,648,DHW)
  float* off2 = off1 + (size_t)NB * COFF * DHW;              // (B,648,DHW)

  // Workspace layout (bf16 col planes dominate; fits in 192MB L2)
  __bf16* colHi = (__bf16*)d_ws;                             // NTOT*KTOT
  __bf16* colLo = colHi + (size_t)NTOT * KTOT;
  __bf16* aHi   = colLo + (size_t)NTOT * KTOT;               // MPAD*KTOT
  __bf16* aLo   = aHi + (size_t)MPAD * KTOT;
  float*  ybuf  = (float*)(aLo + (size_t)MPAD * KTOT);       // (B,64,DHW)
  float*  ybuf2 = ybuf + (size_t)NB * CIN * DHW;
  float*  stats = ybuf2 + (size_t)NB * CIN * DHW;            // 2*64

  const int elems_col   = NTOT * KTOT;                       // 28,311,552
  const int elems_wpad  = MPAD * KTOT;                       // 1,161,216
  const int elems_wdef  = CIN * KTOT;                        //   110,592
  const int elems_defrm = NB * 8 * 27 * DHW;                 // 3,538,944
  const int elems_y     = NB * CIN * DHW;                    // 1,048,576

  dim3 gemmOffGrid(MPAD / 32, NTOT / 32);                    // 21 x 512
  dim3 gemmDefGrid(CIN / 32, NTOT / 32);                     //  2 x 512

  // ---- im2col of x (shared by both offset convs)
  im2col_x_kernel<<<(elems_col + 255) / 256, 256, 0, stream>>>(x, colHi, colLo);

  // ---- offset conv 1 -> off1 (fp32, consumed by deform stage 1)
  split_w_kernel<<<(elems_wpad + 255) / 256, 256, 0, stream>>>(w_off1, aHi, aLo, COFF, MPAD);
  wmma_gemm_kernel<<<gemmOffGrid, 32, 0, stream>>>(aHi, aLo, colHi, colLo, b_off1, off1, COFF);

  // ---- offset conv 2 -> off2 (same col buffer, new weights)
  split_w_kernel<<<(elems_wpad + 255) / 256, 256, 0, stream>>>(w_off2, aHi, aLo, COFF, MPAD);
  wmma_gemm_kernel<<<gemmOffGrid, 32, 0, stream>>>(aHi, aLo, colHi, colLo, b_off2, off2, COFF);

  // ---- deformable conv 1: sample x with off1, GEMM with w1
  deform_im2col_kernel<<<(elems_defrm + 255) / 256, 256, 0, stream>>>(x, off1, colHi, colLo);
  split_w_kernel<<<(elems_wdef + 255) / 256, 256, 0, stream>>>(w1, aHi, aLo, CIN, CIN);
  wmma_gemm_kernel<<<gemmDefGrid, 32, 0, stream>>>(aHi, aLo, colHi, colLo, nullptr, ybuf, CIN);

  // ---- BN1 + ReLU
  bn_stats_kernel<<<CIN, 256, 0, stream>>>(ybuf, stats);
  bn_apply_kernel<<<(elems_y + 255) / 256, 256, 0, stream>>>(ybuf, stats, g1, be1, nullptr, ybuf2, 1);

  // ---- deformable conv 2: sample ybuf2 with off2, GEMM with w2
  deform_im2col_kernel<<<(elems_defrm + 255) / 256, 256, 0, stream>>>(ybuf2, off2, colHi, colLo);
  split_w_kernel<<<(elems_wdef + 255) / 256, 256, 0, stream>>>(w2, aHi, aLo, CIN, CIN);
  wmma_gemm_kernel<<<gemmDefGrid, 32, 0, stream>>>(aHi, aLo, colHi, colLo, nullptr, ybuf, CIN);

  // ---- BN2 + residual(x) + ReLU -> final output
  bn_stats_kernel<<<CIN, 256, 0, stream>>>(ybuf, stats);
  bn_apply_kernel<<<(elems_y + 255) / 256, 256, 0, stream>>>(ybuf, stats, g2, be2, x, out, 1);
}
